// MLP_256_63436666961937
// MI455X (gfx1250) — matrix operand order repaired and checked by reading, not compiled, not run on hardware
//
#include <hip/hip_runtime.h>

typedef __bf16 v16bf __attribute__((ext_vector_type(16)));
typedef float  v8f   __attribute__((ext_vector_type(8)));

#define EPSBN 1e-5f
#define HW2    262144          // 512*512
#define NPIX   2097152         // 8*512*512
#define NTILES 131072          // NPIX/16
#define MBLOCKS 2048
#define WPB 8
#define TOTALW (MBLOCKS*WPB)   // 16384 waves -> 8 tiles/wave

// d_ws layout (bytes): pre-swizzled bf16 A-fragments + folded biases
#define OFF_A1 0               // 8  tiles * 1024B  (Co=128, Kt=1)
#define OFF_A2 8192            // 16 tiles          (Co=64,  Kt=4)
#define OFF_A3 24576           // 4  tiles          (Co=32,  Kt=2)
#define OFF_A4 28672           // 1  tile           (Co=16,  Kt=1)
#define OFF_BF 29696           // 257 f32: b1(128) b2(64) b3(32) b4(16) wh(16) bh(1)

union Frag  { v16bf v; uint4 q[2]; };
union Pack8 { __bf16 h[8]; uint4 q; };
union AccU  { v8f v; float4 f[2]; };

// ---------------------------------------------------------------------------
// Setup: fold BN into weights, convert to bf16, swizzle into this kernel's A-fragment
// lane layout (16-bit A 16x32: lane holds M=co, VGPR v holds packed K pair,
// k_local = 16h+2v+s, h = lane>>4; contiguous k order, B fragments use the same).
// ---------------------------------------------------------------------------
__global__ void pack_kernel(
    const float* w1, const float* g1, const float* b1, const float* m1, const float* v1,
    const float* w2, const float* g2, const float* b2, const float* m2, const float* v2,
    const float* w3, const float* g3, const float* b3, const float* m3, const float* v3,
    const float* w4, const float* g4, const float* b4, const float* m4, const float* v4,
    const float* wh, const float* bh, unsigned char* ws)
{
  int t = blockIdx.x * blockDim.x + threadIdx.x;
  if (t < 29 * 32) {
    int tile = t >> 5, lane = t & 31;
    int ln = lane & 15, lh = lane >> 4;
    const float *w, *g, *v; int Ci, mt, kt, base;
    if (tile < 8)       { w=w1; g=g1; v=v1; Ci=3;   mt=tile;          kt=0;            base=OFF_A1+tile*1024; }
    else if (tile < 24) { int u=tile-8;  w=w2; g=g2; v=v2; Ci=128; mt=u>>2; kt=u&3;   base=OFF_A2+u*1024; }
    else if (tile < 28) { int u=tile-24; w=w3; g=g3; v=v3; Ci=64;  mt=u>>1; kt=u&1;   base=OFF_A3+u*1024; }
    else                { w=w4; g=g4; v=v4; Ci=32;  mt=0;  kt=0;                      base=OFF_A4; }
    int co = mt * 16 + ln;
    float inv = g[co] * (1.0f / sqrtf(v[co] + EPSBN));
    __bf16 vals[16];
    #pragma unroll
    for (int vv = 0; vv < 8; ++vv)
      #pragma unroll
      for (int s = 0; s < 2; ++s) {
        int kl = 16*lh + 2*vv + s;
        int ci = kt * 32 + kl;
        float val = (ci < Ci) ? w[co * Ci + ci] * inv : 0.0f;
        vals[vv*2 + s] = (__bf16)val;
      }
    uint4* dst = (uint4*)(ws + base + lane * 32);
    dst[0] = ((uint4*)vals)[0];
    dst[1] = ((uint4*)vals)[1];
  } else if (t < 29*32 + 257) {
    int u = t - 29*32;
    float* f = (float*)(ws + OFF_BF);
    if (u < 128)      { float inv=g1[u]*(1.0f/sqrtf(v1[u]+EPSBN));              f[u] = b1[u] - m1[u]*inv; }
    else if (u < 192) { int c=u-128; float inv=g2[c]*(1.0f/sqrtf(v2[c]+EPSBN)); f[u] = b2[c] - m2[c]*inv; }
    else if (u < 224) { int c=u-192; float inv=g3[c]*(1.0f/sqrtf(v3[c]+EPSBN)); f[u] = b3[c] - m3[c]*inv; }
    else if (u < 240) { int c=u-224; float inv=g4[c]*(1.0f/sqrtf(v4[c]+EPSBN)); f[u] = b4[c] - m4[c]*inv; }
    else if (u < 256) { f[u] = wh[u - 240]; }
    else              { f[u] = bh[0]; }
  }
}

// ---------------------------------------------------------------------------
// Fused MLP: per wave, 16 pixels at a time. A=weights(co x ci), B=acts(ci x pix).
// Folded bias rides in as the WMMA C operand. D layout -> each lane holds 8
// consecutive co for its pixel -> one b128 LDS store; next layer's B fragment
// is contiguous b128 LDS loads. 29 bf16 WMMAs per tile.
// ---------------------------------------------------------------------------
__global__ __launch_bounds__(256) void mlp_kernel(
    const float* __restrict__ x, const unsigned char* __restrict__ ws,
    float* __restrict__ out)
{
  __shared__ __align__(16) unsigned char smW[29696];
  __shared__ __align__(16) float smB[260];
  // per-wave staging: actA (L1 out, pitch 272B; reused for L3 out, pitch 80B)
  //                   actB (L2 out, pitch 144B)
  __shared__ __align__(16) unsigned char smAct[WPB][4352 + 2304];

  {
    const uint4* src = (const uint4*)ws;
    uint4* dst = (uint4*)smW;
    for (int i = threadIdx.x; i < 29696/16; i += blockDim.x) dst[i] = src[i];
    const float* bs = (const float*)(ws + OFF_BF);
    for (int i = threadIdx.x; i < 257; i += blockDim.x) smB[i] = bs[i];
  }
  __syncthreads();

  const int lane = threadIdx.x & 31;
  const int ln = lane & 15, lh = lane >> 4;
  const int gwave = blockIdx.x * WPB + (threadIdx.x >> 5);

  unsigned char* actA = smAct[threadIdx.x >> 5];
  unsigned char* actB = actA + 4352;

  const float* biasL1 = smB;
  const float* biasL2 = smB + 128;
  const float* biasL3 = smB + 192;
  const float* biasL4 = smB + 224;

  // ---- loop-invariant register-resident fragments ----
  float hwv[8];
  #pragma unroll
  for (int r = 0; r < 8; ++r) hwv[r] = smB[240 + 8*lh + r];
  const float bhv = smB[256];

  Frag a3[4];
  #pragma unroll
  for (int i = 0; i < 4; ++i) {
    const uint4* ap = (const uint4*)(smW + OFF_A3 + i*1024 + lane*32);
    a3[i].q[0] = ap[0]; a3[i].q[1] = ap[1];
  }
  Frag a4;
  {
    const uint4* ap = (const uint4*)(smW + OFF_A4 + lane*32);
    a4.q[0] = ap[0]; a4.q[1] = ap[1];
  }
  AccU c3[2];
  #pragma unroll
  for (int i = 0; i < 2; ++i) {
    const float4* bp = (const float4*)(biasL3 + i*16 + 8*lh);
    c3[i].f[0] = bp[0]; c3[i].f[1] = bp[1];
  }
  AccU c4;
  {
    const float4* bp = (const float4*)(biasL4 + 8*lh);
    c4.f[0] = bp[0]; c4.f[1] = bp[1];
  }

  for (int tile = gwave; tile < NTILES; tile += TOTALW) {
    const int p0 = tile << 4;
    const int n  = p0 >> 18;
    const int hw = p0 & (HW2 - 1);
    const float* xb = x + (size_t)n * 3 * HW2 + hw + ln;
    const float x0 = xb[0], x1 = xb[HW2], x2 = xb[2 * HW2];

    // prefetch next tile's x (speculative, L2 pull)
    {
      int tn = tile + TOTALW;
      if (tn < NTILES) {
        int pp = tn << 4;
        const float* px = x + (size_t)(pp >> 18) * 3 * HW2 + (pp & (HW2-1)) + ln;
        __builtin_prefetch(px, 0, 0);
        __builtin_prefetch(px + HW2, 0, 0);
        __builtin_prefetch(px + 2*HW2, 0, 0);
      }
    }

    // B fragment for layer 1: K = ci (0..2 live, padded to 32), N = pixel
    Frag bf1;
    bf1.q[0] = make_uint4(0,0,0,0); bf1.q[1] = make_uint4(0,0,0,0);
    const __bf16 z = (__bf16)0.0f;
    bf1.v[0] = lh ? z : (__bf16)x0;
    bf1.v[1] = lh ? z : (__bf16)x1;
    bf1.v[2] = lh ? z : (__bf16)x2;

    // ---- Layer 1: 3 -> 128 (bias as C) ----
    #pragma unroll
    for (int mt = 0; mt < 8; ++mt) {
      Frag a;
      const uint4* ap = (const uint4*)(smW + OFF_A1 + mt*1024 + lane*32);
      a.q[0] = ap[0]; a.q[1] = ap[1];
      AccU acc;
      const float4* cp = (const float4*)(biasL1 + mt*16 + 8*lh);
      acc.f[0] = cp[0]; acc.f[1] = cp[1];
      acc.v = __builtin_amdgcn_wmma_f32_16x16x32_bf16(false, a.v, false, bf1.v,
                                                      (short)0, acc.v, false, false);
      Pack8 pk;
      #pragma unroll
      for (int r = 0; r < 8; ++r)
        pk.h[r] = (__bf16)fmaxf(acc.v[r], 0.0f);
      *(uint4*)(actA + ln*272 + mt*32 + 16*lh) = pk.q;
    }
    asm volatile("s_wait_dscnt 0" ::: "memory");

    // ---- Layer 2: 128 -> 64 ----
    Frag b2[4];
    #pragma unroll
    for (int kt = 0; kt < 4; ++kt) {
      const uint4* bp = (const uint4*)(actA + ln*272 + kt*64 + 32*lh);
      b2[kt].q[0] = bp[0]; b2[kt].q[1] = bp[1];
    }
    #pragma unroll
    for (int mt = 0; mt < 4; ++mt) {
      AccU acc;
      const float4* cp = (const float4*)(biasL2 + mt*16 + 8*lh);
      acc.f[0] = cp[0]; acc.f[1] = cp[1];
      #pragma unroll
      for (int kt = 0; kt < 4; ++kt) {
        Frag a;
        const uint4* ap = (const uint4*)(smW + OFF_A2 + (mt*4+kt)*1024 + lane*32);
        a.q[0] = ap[0]; a.q[1] = ap[1];
        acc.v = __builtin_amdgcn_wmma_f32_16x16x32_bf16(false, a.v, false, b2[kt].v,
                                                        (short)0, acc.v, false, false);
      }
      Pack8 pk;
      #pragma unroll
      for (int r = 0; r < 8; ++r)
        pk.h[r] = (__bf16)fmaxf(acc.v[r], 0.0f);
      *(uint4*)(actB + ln*144 + mt*32 + 16*lh) = pk.q;
    }
    asm volatile("s_wait_dscnt 0" ::: "memory");

    // ---- Layer 3: 64 -> 32 (A-frags + bias resident in registers) ----
    Frag b3[2];
    #pragma unroll
    for (int kt = 0; kt < 2; ++kt) {
      const uint4* bp = (const uint4*)(actB + ln*144 + kt*64 + 32*lh);
      b3[kt].q[0] = bp[0]; b3[kt].q[1] = bp[1];
    }
    #pragma unroll
    for (int mt = 0; mt < 2; ++mt) {
      AccU acc;
      acc.v = c3[mt].v;
      #pragma unroll
      for (int kt = 0; kt < 2; ++kt) {
        acc.v = __builtin_amdgcn_wmma_f32_16x16x32_bf16(false, a3[mt*2+kt].v, false, b3[kt].v,
                                                        (short)0, acc.v, false, false);
      }
      Pack8 pk;
      #pragma unroll
      for (int r = 0; r < 8; ++r)
        pk.h[r] = (__bf16)fmaxf(acc.v[r], 0.0f);
      *(uint4*)(actA + ln*80 + mt*32 + 16*lh) = pk.q;   // reuse actA, pitch 80B
    }
    asm volatile("s_wait_dscnt 0" ::: "memory");

    // ---- Layer 4: 32 -> 16 (A-frag + bias resident) ----
    Frag b4;
    {
      const uint4* bp = (const uint4*)(actA + ln*80 + 32*lh);
      b4.q[0] = bp[0]; b4.q[1] = bp[1];
    }
    AccU acc;
    acc.v = c4.v;
    acc.v = __builtin_amdgcn_wmma_f32_16x16x32_bf16(false, a4.v, false, b4.v,
                                                    (short)0, acc.v, false, false);

    // ---- Head: 16 -> 1 (per-lane dot over 8 co + cross-half reduce) ----
    float partial = 0.0f;
    #pragma unroll
    for (int r = 0; r < 8; ++r)
      partial += hwv[r] * fmaxf(acc.v[r], 0.0f);
    partial += __shfl_xor(partial, 16, 32);
    const float res = partial + bhv;
    if (lane < 16) out[p0 + ln] = res;
  }
}

// ---------------------------------------------------------------------------
extern "C" void kernel_launch(void* const* d_in, const int* in_sizes, int n_in,
                              void* d_out, int out_size, void* d_ws, size_t ws_size,
                              hipStream_t stream) {
  (void)in_sizes; (void)n_in; (void)out_size; (void)ws_size;
  const float* x  = (const float*)d_in[0];
  const float* w1 = (const float*)d_in[1];  const float* g1 = (const float*)d_in[2];
  const float* b1 = (const float*)d_in[3];  const float* m1 = (const float*)d_in[4];
  const float* v1 = (const float*)d_in[5];
  const float* w2 = (const float*)d_in[6];  const float* g2 = (const float*)d_in[7];
  const float* b2 = (const float*)d_in[8];  const float* m2 = (const float*)d_in[9];
  const float* v2 = (const float*)d_in[10];
  const float* w3 = (const float*)d_in[11]; const float* g3 = (const float*)d_in[12];
  const float* b3 = (const float*)d_in[13]; const float* m3 = (const float*)d_in[14];
  const float* v3 = (const float*)d_in[15];
  const float* w4 = (const float*)d_in[16]; const float* g4 = (const float*)d_in[17];
  const float* b4 = (const float*)d_in[18]; const float* m4 = (const float*)d_in[19];
  const float* v4 = (const float*)d_in[20];
  const float* wh = (const float*)d_in[21]; const float* bh = (const float*)d_in[22];
  unsigned char* ws = (unsigned char*)d_ws;

  pack_kernel<<<5, 256, 0, stream>>>(w1,g1,b1,m1,v1, w2,g2,b2,m2,v2,
                                     w3,g3,b3,m3,v3, w4,g4,b4,m4,v4,
                                     wh, bh, ws);
  mlp_kernel<<<MBLOCKS, 256, 0, stream>>>(x, ws, (float*)d_out);
}
